// LSTMRecurrentLayer_10333691314206
// MI455X (gfx1250) — matrix operand order repaired and checked by reading, not compiled, not run on hardware
//
#include <hip/hip_runtime.h>

// ---------------------------------------------------------------------------
// LSTM on MI455X (gfx1250): fused [h|x] x [W_hh|W_ih]^T recurrence using
// v_wmma_f32_16x16x32_bf16, fp32 accumulation, ping-pong hidden state.
// ---------------------------------------------------------------------------

typedef __attribute__((ext_vector_type(16))) __bf16 v16bf;
typedef __attribute__((ext_vector_type(8)))  float  v8f;

#define T_STEPS 512
#define BATCH   256
#define DIN     128
#define HID     256
#define KTOT    384   // HID + DIN (concatenated K)
#define G4      1024  // 4 * HID

struct __align__(16) U4 { unsigned x, y, z, w; };

union Frag {
    struct { U4 lo, hi; } u;
    v16bf v;
    __device__ Frag() {}
};

// Load a 16-element bf16 WMMA fragment as two 16B chunks, `gap` elements apart.
__device__ inline v16bf load_frag(const __bf16* p, int gap) {
    Frag f;
    f.u.lo = *(const U4*)(p);
    f.u.hi = *(const U4*)(p + gap);
    return f.v;
}

__device__ inline float fsig(float x)  { return 1.0f / (1.0f + __expf(-x)); }
__device__ inline float ftanh(float x) { return 1.0f - 2.0f / (1.0f + __expf(2.0f * x)); }

// ---------------------------------------------------------------------------
// One LSTM timestep:
//   gates[B,4H] = [h|x][B,384] @ Wcat[1024,384]^T + bias
//   i,f,g,o -> c,h update. Each wave owns a 16(b) x 16(j) tile and its 4 gate
//   columns, so the c update is race-free; h is double-buffered across steps.
// grid = (16 j-tiles, 4), block = 128 (4 waves, wave w -> b-tile blockIdx.y*4+w)
// ---------------------------------------------------------------------------
__global__ __launch_bounds__(128) void lstm_step_kernel(
    const __bf16* __restrict__ xt,     // [B, DIN] bf16 (this timestep)
    const __bf16* __restrict__ Wcat,   // [1024, 384] bf16, row-major
    const float*  __restrict__ bias,   // [1024] = b_ih + b_hh
    const __bf16* __restrict__ h_in,   // [B, HID] bf16
    __bf16*       __restrict__ h_out,  // [B, HID] bf16
    float*        __restrict__ c,      // [B, HID] fp32 (in-place)
    float*        __restrict__ out_t)  // [B, HID] fp32 output slice
{
    const int lane = threadIdx.x & 31;
    const int wave = threadIdx.x >> 5;
    const int hf   = lane >> 4;    // lane half (0/1)
    const int mr   = lane & 15;    // row within A tile / col within B,C,D tile

    const int b0 = (blockIdx.y * 4 + wave) * 16;
    const int j0 = blockIdx.x * 16;

    // A-matrix rows: [h_in row | x row] for batch row (b0+mr)
    const __bf16* hrow = h_in + (size_t)(b0 + mr) * HID;
    const __bf16* xrow = xt   + (size_t)(b0 + mr) * DIN;

    // B-matrix: column n = Wcat row (gate*HID + j0 + n), K along the row.
    const __bf16* wr0 = Wcat + (size_t)(0 * HID + j0 + mr) * KTOT;
    const __bf16* wr1 = Wcat + (size_t)(1 * HID + j0 + mr) * KTOT;
    const __bf16* wr2 = Wcat + (size_t)(2 * HID + j0 + mr) * KTOT;
    const __bf16* wr3 = Wcat + (size_t)(3 * HID + j0 + mr) * KTOT;

    v8f acc0 = {}; v8f acc1 = {}; v8f acc2 = {}; v8f acc3 = {};

    // K = 0..255 from h_in (8 k-steps of 32)
    #pragma unroll
    for (int ks = 0; ks < 8; ++ks) {
        const int k0 = ks * 32;
        v16bf a = load_frag(hrow + k0 + hf * 16, 8);
        const int kb = k0 + hf * 16;
        v16bf b0f = load_frag(wr0 + kb, 8);
        v16bf b1f = load_frag(wr1 + kb, 8);
        v16bf b2f = load_frag(wr2 + kb, 8);
        v16bf b3f = load_frag(wr3 + kb, 8);
        acc0 = __builtin_amdgcn_wmma_f32_16x16x32_bf16(false, a, false, b0f, (short)0, acc0, false, false);
        acc1 = __builtin_amdgcn_wmma_f32_16x16x32_bf16(false, a, false, b1f, (short)0, acc1, false, false);
        acc2 = __builtin_amdgcn_wmma_f32_16x16x32_bf16(false, a, false, b2f, (short)0, acc2, false, false);
        acc3 = __builtin_amdgcn_wmma_f32_16x16x32_bf16(false, a, false, b3f, (short)0, acc3, false, false);
    }
    // K = 256..383 from x_t (4 k-steps of 32)
    #pragma unroll
    for (int ks = 0; ks < 4; ++ks) {
        const int k0 = ks * 32;
        v16bf a = load_frag(xrow + k0 + hf * 16, 8);
        const int kb = HID + k0 + hf * 16;
        v16bf b0f = load_frag(wr0 + kb, 8);
        v16bf b1f = load_frag(wr1 + kb, 8);
        v16bf b2f = load_frag(wr2 + kb, 8);
        v16bf b3f = load_frag(wr3 + kb, 8);
        acc0 = __builtin_amdgcn_wmma_f32_16x16x32_bf16(false, a, false, b0f, (short)0, acc0, false, false);
        acc1 = __builtin_amdgcn_wmma_f32_16x16x32_bf16(false, a, false, b1f, (short)0, acc1, false, false);
        acc2 = __builtin_amdgcn_wmma_f32_16x16x32_bf16(false, a, false, b2f, (short)0, acc2, false, false);
        acc3 = __builtin_amdgcn_wmma_f32_16x16x32_bf16(false, a, false, b3f, (short)0, acc3, false, false);
    }

    // LSTM cell epilogue on the accumulator layout:
    // VGPR r, lane -> element (row = r + 8*half, col = lane&15) of the tile.
    const float bi = bias[0 * HID + j0 + mr];
    const float bf = bias[1 * HID + j0 + mr];
    const float bg = bias[2 * HID + j0 + mr];
    const float bo = bias[3 * HID + j0 + mr];

    #pragma unroll
    for (int r = 0; r < 8; ++r) {
        const int row = b0 + r + hf * 8;
        const size_t idx = (size_t)row * HID + (j0 + mr);
        const float iv = fsig(acc0[r] + bi);
        const float fv = fsig(acc1[r] + bf);
        const float gv = ftanh(acc2[r] + bg);
        const float ov = fsig(acc3[r] + bo);
        const float cn = fv * c[idx] + iv * gv;
        c[idx] = cn;
        const float hv = ov * ftanh(cn);
        out_t[idx] = hv;
        h_out[idx] = (__bf16)hv;
    }
}

// ---------------------------------------------------------------------------
// Prologue kernels (run once per call; bandwidth-trivial at 23.3 TB/s)
// ---------------------------------------------------------------------------
__global__ void convert_x_kernel(const float* __restrict__ x,
                                 __bf16* __restrict__ xb, int n) {
    int i = blockIdx.x * blockDim.x + threadIdx.x;
    if (i < n) xb[i] = (__bf16)x[i];
}

// Wcat[row, 0:256] = W_hh[row,:], Wcat[row, 256:384] = W_ih[row,:]
__global__ void pack_w_kernel(const float* __restrict__ Wih,
                              const float* __restrict__ Whh,
                              __bf16* __restrict__ Wcat) {
    int i = blockIdx.x * blockDim.x + threadIdx.x;   // over 1024*384
    int row = i / KTOT, col = i % KTOT;
    float v = (col < HID) ? Whh[row * HID + col] : Wih[row * DIN + (col - HID)];
    Wcat[i] = (__bf16)v;
}

// Zero h ping-pong buffers + c state, fold biases (workspace is poisoned).
__global__ void init_kernel(const float* __restrict__ bih,
                            const float* __restrict__ bhh,
                            float* __restrict__ bias,
                            __bf16* __restrict__ h0, __bf16* __restrict__ h1,
                            float* __restrict__ c) {
    int i = blockIdx.x * blockDim.x + threadIdx.x;   // over BATCH*HID
    h0[i] = (__bf16)0.0f;
    h1[i] = (__bf16)0.0f;
    c[i]  = 0.0f;
    if (i < G4) bias[i] = bih[i] + bhh[i];
}

// ---------------------------------------------------------------------------
// Workspace layout (bytes):
//   x_bf16 [T,B,DIN]       :        0 .. 33554432
//   Wcat   [1024,384] bf16 : 33554432 ..   786432
//   bias   [1024] f32      : 34340864 ..     4096
//   h0     [B,HID] bf16    : 34344960 ..   131072
//   h1     [B,HID] bf16    : 34476032 ..   131072
//   c      [B,HID] f32     : 34607104 ..   262144   (total ~34.9 MB)
// ---------------------------------------------------------------------------
extern "C" void kernel_launch(void* const* d_in, const int* in_sizes, int n_in,
                              void* d_out, int out_size, void* d_ws, size_t ws_size,
                              hipStream_t stream) {
    (void)in_sizes; (void)n_in; (void)out_size; (void)ws_size;

    const float* inputs = (const float*)d_in[0];   // [T,B,DIN]
    const float* Wih    = (const float*)d_in[1];   // [4H, DIN]
    const float* Whh    = (const float*)d_in[2];   // [4H, HID]
    const float* bih    = (const float*)d_in[3];   // [4H]
    const float* bhh    = (const float*)d_in[4];   // [4H]
    float* out = (float*)d_out;                    // [T,B,HID]

    char* ws = (char*)d_ws;
    __bf16* xb   = (__bf16*)(ws + 0);
    __bf16* Wcat = (__bf16*)(ws + 33554432);
    float*  bias = (float*) (ws + 34340864);
    __bf16* h0   = (__bf16*)(ws + 34344960);
    __bf16* h1   = (__bf16*)(ws + 34476032);
    float*  cst  = (float*) (ws + 34607104);

    const int nx = T_STEPS * BATCH * DIN;           // 16,777,216
    convert_x_kernel<<<nx / 256, 256, 0, stream>>>(inputs, xb, nx);
    pack_w_kernel<<<(G4 * KTOT) / 256, 256, 0, stream>>>(Wih, Whh, Wcat);
    init_kernel<<<(BATCH * HID) / 256, 256, 0, stream>>>(bih, bhh, bias, h0, h1, cst);

    for (int t = 0; t < T_STEPS; ++t) {
        const __bf16* hi = (t & 1) ? h1 : h0;
        __bf16*       ho = (t & 1) ? h0 : h1;
        lstm_step_kernel<<<dim3(16, 4), 128, 0, stream>>>(
            xb + (size_t)t * BATCH * DIN, Wcat, bias, hi, ho, cst,
            out + (size_t)t * BATCH * HID);
    }
}
